// SparseMaskExtraction_46686294507604
// MI455X (gfx1250) — matrix operand order repaired and checked by reading, not compiled, not run on hardware
//
#include <hip/hip_runtime.h>
#include <hip/hip_bf16.h>

// Problem constants (match reference)
#define BB   4
#define SS   1024
#define TT   1024
#define DDIM 256
#define HH   16
#define DHH  64
#define VDD  1024

typedef __attribute__((ext_vector_type(16))) __bf16 v16bf;
typedef __attribute__((ext_vector_type(8)))  __bf16 v8bf;
typedef __attribute__((ext_vector_type(8)))  float  v8f;
typedef __attribute__((ext_vector_type(4))) unsigned int u32x4;
typedef __attribute__((ext_vector_type(8))) int          i32x8;
typedef __attribute__((ext_vector_type(4))) int          i32x4;

#ifndef __has_builtin
#define __has_builtin(x) 0
#endif
#if __has_builtin(__builtin_amdgcn_tensor_load_to_lds) && __has_builtin(__builtin_amdgcn_s_wait_tensorcnt)
#define HAVE_TDM 1
#else
#define HAVE_TDM 0
#endif

static __device__ inline v8f vzero8() {
    v8f r;
#pragma unroll
    for (int i = 0; i < 8; ++i) r[i] = 0.f;
    return r;
}

// Load a 16-element bf16 fragment from two 16-byte-aligned 8-element chunks.
static __device__ inline v16bf load_ab(const __bf16* p0, const __bf16* p1) {
    v8bf lo = *reinterpret_cast<const v8bf*>(p0);
    v8bf hi = *reinterpret_cast<const v8bf*>(p1);
    return __builtin_shufflevector(lo, hi, 0,1,2,3,4,5,6,7,8,9,10,11,12,13,14,15);
}
// Load a contiguous 32-byte-aligned 16-element bf16 fragment.
static __device__ inline v16bf load_bfrag(const __bf16* p) {
    return *reinterpret_cast<const v16bf*>(p);
}

static __device__ inline v8f wmma_bf16(v16bf a, v16bf b, v8f c) {
    return __builtin_amdgcn_wmma_f32_16x16x32_bf16(false, a, false, b, (short)0, c, false, false);
}

static __device__ inline float silu_f(float x) { return x / (1.f + __expf(-x)); }

#if HAVE_TDM
// Issue TENSOR_LOAD_TO_LDS for a 2D bf16 tile: rows of length tileK (elements),
// tileRows rows, source row stride = K elements, dest LDS compact (tileK-stride).
// D# layout per CDNA5 ISA ch.8 (group0 128b, group1 256b; groups 2/3 unused->0).
static __device__ inline void tdm_load_tile_2d(unsigned lds_off, const void* gaddr,
                                               int K, int rowsTotal, int tileK, int tileRows) {
    unsigned long long ga = (unsigned long long)(uintptr_t)gaddr;
    u32x4 g0;
    g0[0] = 1u;                                        // count=1, user descriptor
    g0[1] = lds_off;                                   // lds_addr (bytes)
    g0[2] = (unsigned)(ga & 0xffffffffu);              // global_addr[31:0]
    g0[3] = (unsigned)((ga >> 32) & 0x01ffffffu) | (2u << 30);  // addr[56:32] | type=2
    unsigned long long d0s = (unsigned long long)(unsigned)K;   // tensor_dim0_stride
    i32x8 g1;
    g1[0] = (int)(1u << 16);                           // workgroup_mask=0, data_size=1 (2B)
    g1[1] = (int)(((unsigned)K & 0xffffu) << 16);      // tensor_dim0[15:0] @ bits 63:48
    g1[2] = (int)((((unsigned)K >> 16) & 0xffffu) |    // tensor_dim0[31:16]
                  (((unsigned)rowsTotal & 0xffffu) << 16));      // tensor_dim1[15:0]
    g1[3] = (int)((((unsigned)rowsTotal >> 16) & 0xffffu) |      // tensor_dim1[31:16]
                  (((unsigned)tileK & 0xffffu) << 16));          // tile_dim0
    g1[4] = (int)((unsigned)tileRows & 0xffffu);       // tile_dim1 (tile_dim2 = 0)
    g1[5] = (int)(d0s & 0xffffffffu);                  // tensor_dim0_stride[31:0]
    g1[6] = (int)((d0s >> 32) & 0xffffu);              // stride[47:32]; dim1_stride lo = 0
    g1[7] = 0;
    i32x4 gz = {0, 0, 0, 0};
#if __clang_major__ >= 23
    i32x8 gz8 = {0, 0, 0, 0, 0, 0, 0, 0};
    __builtin_amdgcn_tensor_load_to_lds(g0, g1, gz, gz, gz8, 0);
#else
    __builtin_amdgcn_tensor_load_to_lds(g0, g1, gz, gz, 0);
#endif
}
#endif  // HAVE_TDM

// ---------------------------------------------------------------------------
// fp32 -> bf16 converter (weights)
// ---------------------------------------------------------------------------
__global__ void f32_to_bf16(const float* __restrict__ src, __bf16* __restrict__ dst, int n) {
    int i = blockIdx.x * blockDim.x + threadIdx.x;
    if (i < n) dst[i] = (__bf16)src[i];
}

// ---------------------------------------------------------------------------
// LayerNorm over rows of width 256, fp32 in -> bf16 out. One wave per row.
// ---------------------------------------------------------------------------
__global__ __launch_bounds__(256) void ln_rows(const float* __restrict__ X,
                                               const float* __restrict__ g,
                                               const float* __restrict__ b,
                                               __bf16* __restrict__ Y, int rows) {
    int w = (blockIdx.x * blockDim.x + threadIdx.x) >> 5;
    int lane = threadIdx.x & 31;
    if (w >= rows) return;
    const float* xr = X + (size_t)w * DDIM;
    float v[8];
    float s = 0.f;
#pragma unroll
    for (int i = 0; i < 8; ++i) { v[i] = xr[i * 32 + lane]; s += v[i]; }
#pragma unroll
    for (int m = 16; m >= 1; m >>= 1) s += __shfl_xor(s, m, 32);
    float mean = s * (1.f / DDIM);
    float ss = 0.f;
#pragma unroll
    for (int i = 0; i < 8; ++i) { float d = v[i] - mean; ss += d * d; }
#pragma unroll
    for (int m = 16; m >= 1; m >>= 1) ss += __shfl_xor(ss, m, 32);
    float rstd = rsqrtf(ss * (1.f / DDIM) + 1e-5f);
    __bf16* yr = Y + (size_t)w * DDIM;
#pragma unroll
    for (int i = 0; i < 8; ++i) {
        int idx = i * 32 + lane;
        yr[idx] = (__bf16)((v[i] - mean) * rstd * g[idx] + b[idx]);
    }
}

// ---------------------------------------------------------------------------
// Tiny coordinate MLP: in[.,2] (optionally minus per-batch pos) -> silu(L1) ->
// silu(L2) -> out32 bf16 [rows,32]. One thread per row.
// ---------------------------------------------------------------------------
__global__ void mlp2_small(const float* __restrict__ in2, const float* __restrict__ pos,
                           const float* __restrict__ W1, const float* __restrict__ b1,
                           const float* __restrict__ W2, const float* __restrict__ b2,
                           __bf16* __restrict__ out32, int rows, int rows_per_b) {
    int r = blockIdx.x * blockDim.x + threadIdx.x;
    if (r >= rows) return;
    float a = in2[2 * r], c = in2[2 * r + 1];
    if (pos) {
        int bb = r / rows_per_b;
        a -= pos[2 * bb];
        c -= pos[2 * bb + 1];
    }
    float h1[32];
#pragma unroll
    for (int j = 0; j < 32; ++j) {
        float t = W1[2 * j] * a + W1[2 * j + 1] * c + b1[j];
        h1[j] = silu_f(t);
    }
#pragma unroll 4
    for (int j = 0; j < 32; ++j) {
        float t = b2[j];
        for (int q2 = 0; q2 < 32; ++q2) t += W2[j * 32 + q2] * h1[q2];
        out32[(size_t)r * 32 + j] = (__bf16)silu_f(t);
    }
}

// ---------------------------------------------------------------------------
// One KC-chunk of the 16x64 per-wave WMMA tile; B fragments read from LDS.
// ---------------------------------------------------------------------------
static __device__ inline void gemm_chunk(const __bf16* __restrict__ Arow, const __bf16* wt,
                                         int kbase, int KC, int hl, int lid, v8f acc[4]) {
    for (int k2 = 0; k2 < KC; k2 += 32) {
        __builtin_prefetch(Arow + kbase + k2 + 128, 0, 1);
        v16bf a = load_ab(Arow + kbase + k2 + hl * 16, Arow + kbase + k2 + hl * 16 + 8);
#pragma unroll
        for (int s = 0; s < 4; ++s) {
            v16bf bf = load_bfrag(wt + (size_t)(s * 16 + lid) * KC + k2 + hl * 16);
            acc[s] = wmma_bf16(a, bf, acc[s]);
        }
    }
}

// ---------------------------------------------------------------------------
// Generic WMMA bf16 GEMM:  out = act(A[M,K] x W[N,K]^T + bias) * alpha + resid
// Wave computes 16(M) x 64(N); block = 8 waves -> 128 x 64 tile.
// Weight tile (64 x KC) is staged in LDS: TDM double-buffered async DMA
// (TENSORcnt-tracked) when available, cooperative copy otherwise.
// ---------------------------------------------------------------------------
__global__ __launch_bounds__(256) void gemm_bf16_wmma(
    const __bf16* __restrict__ A, const __bf16* __restrict__ W,
    const float* __restrict__ bias, const float* __restrict__ alpha,
    const float* __restrict__ resid, float* __restrict__ outF,
    __bf16* __restrict__ outB, int M, int N, int K, int ldc, int coloffs, int act_silu) {
    __shared__ alignas(64) __bf16 wtile[2][64 * 128];  // 32 KB double buffer
    const int lane = threadIdx.x & 31;
    const int wv   = threadIdx.x >> 5;
    const int hl   = lane >> 4;   // half-wave id
    const int lid  = lane & 15;
    const int mtile = blockIdx.x * 128 + wv * 16;
    const int nbase = blockIdx.y * 64;
    const int KC = (K % 128 == 0) ? 128 : K;  // K in {32,256,1024,2048}
    const int nc = K / KC;

    const __bf16* Arow = A + (size_t)(mtile + lid) * K;
    v8f acc[4] = {vzero8(), vzero8(), vzero8(), vzero8()};

#if HAVE_TDM
    if (wv == 0)
        tdm_load_tile_2d((unsigned)(uintptr_t)&wtile[0][0],
                         W + (size_t)nbase * K, K, N, KC, 64);
    for (int c = 0; c < nc; ++c) {
        if (wv == 0) {
            if (c + 1 < nc) {
                tdm_load_tile_2d((unsigned)(uintptr_t)&wtile[(c + 1) & 1][0],
                                 W + (size_t)nbase * K + (size_t)(c + 1) * KC, K, N, KC, 64);
                __builtin_amdgcn_s_wait_tensorcnt((short)1);  // chunk c landed
            } else {
                __builtin_amdgcn_s_wait_tensorcnt((short)0);
            }
        }
        __syncthreads();
        gemm_chunk(Arow, &wtile[c & 1][0], c * KC, KC, hl, lid, acc);
        __syncthreads();  // protect buffer (c&1) before it is re-filled
    }
#else
    for (int c = 0; c < nc; ++c) {
        // cooperative staging of W[nbase..+63][c*KC..+KC) into LDS
        const int vecs_per_row = KC / 8;
        for (int t = threadIdx.x; t < 64 * vecs_per_row; t += 256) {
            int row = t / vecs_per_row, col = t % vecs_per_row;
            *(uint4*)(&wtile[0][0] + (size_t)row * KC + col * 8) =
                *(const uint4*)(W + (size_t)(nbase + row) * K + (size_t)c * KC + col * 8);
        }
        __syncthreads();
        gemm_chunk(Arow, &wtile[0][0], c * KC, KC, hl, lid, acc);
        __syncthreads();
    }
#endif

    const float al = alpha ? alpha[0] : 1.0f;
#pragma unroll
    for (int s = 0; s < 4; ++s) {
        int ncol = nbase + s * 16 + lid;
        float bv = bias ? bias[ncol] : 0.f;
#pragma unroll
        for (int v = 0; v < 8; ++v) {
            int row = mtile + v + hl * 8;  // C layout: VGPR v -> M=v (+8 upper half)
            float x = acc[s][v] + bv;
            if (act_silu) x = silu_f(x);
            x *= al;
            size_t oi = (size_t)row * ldc + coloffs + ncol;
            if (resid) x += resid[oi];
            if (outF) outF[oi] = x;
            if (outB) outB[oi] = (__bf16)x;
        }
    }
}

// ---------------------------------------------------------------------------
// v fp32 [B*S, VD] -> vt bf16 [B, H, DH, S]  (V^T per head for PV WMMA)
// ---------------------------------------------------------------------------
__global__ void transpose_v(const float* __restrict__ vf, __bf16* __restrict__ vt) {
    int idx = blockIdx.x * blockDim.x + threadIdx.x;  // ((b*H+h)*DH+d)*S + s
    int s = idx & (SS - 1);
    int d = (idx >> 10) & (DHH - 1);
    int h = (idx >> 16) & (HH - 1);
    int b = idx >> 20;
    vt[idx] = (__bf16)vf[((size_t)(b * SS + s)) * VDD + h * DHH + d];
}

// ---------------------------------------------------------------------------
// Transposed flash attention. One wave per (b, h, 16-query tile).
//   S^T = K x Q^T  (query = lane column -> per-lane softmax state)
//   O^T = V^T x P^T (accumulated over 32-key steps)
// ---------------------------------------------------------------------------
__global__ __launch_bounds__(256) void attn_fwd(const __bf16* __restrict__ q,
                                                const __bf16* __restrict__ k,
                                                const __bf16* __restrict__ vt,
                                                __bf16* __restrict__ o) {
    const int lane = threadIdx.x & 31;
    const int hl   = lane >> 4;
    const int lid  = lane & 15;
    const int gw = (blockIdx.x * blockDim.x + threadIdx.x) >> 5;
    const int ntile = TT / 16;
    if (gw >= BB * HH * ntile) return;
    const int b  = gw / (HH * ntile);
    const int rr = gw % (HH * ntile);
    const int h  = rr / ntile;
    const int tb = (rr % ntile) * 16;

    // Q^T B-fragments: lane = query column; lanes 0-15 hold dh rows 0..7,16..23 of chunk
    const __bf16* qrow = q + ((size_t)(b * TT + tb + lid)) * VDD + h * DHH;
    v16bf bq0 = load_ab(qrow + hl * 8, qrow + 16 + hl * 8);        // dh 0..31
    v16bf bq1 = load_ab(qrow + 32 + hl * 8, qrow + 48 + hl * 8);   // dh 32..63

    v8f acc[4] = {vzero8(), vzero8(), vzero8(), vzero8()};
    float mrun = -1e30f, lrun = 0.f;
    const __bf16* kbp = k + ((size_t)(b * SS)) * VDD + h * DHH;
    const __bf16* vtb = vt + ((size_t)(b * HH + h)) * DHH * SS;
    const float scale = 0.125f;  // 1/sqrt(64)

    for (int sb = 0; sb < SS; sb += 32) {
        v8f sc0 = vzero8(), sc1 = vzero8();
        {   // keys sb..sb+15
            const __bf16* kr = kbp + (size_t)(sb + lid) * VDD;
            sc0 = wmma_bf16(load_ab(kr + hl * 8,      kr + 16 + hl * 8), bq0, sc0);
            sc0 = wmma_bf16(load_ab(kr + 32 + hl * 8, kr + 48 + hl * 8), bq1, sc0);
        }
        {   // keys sb+16..sb+31
            const __bf16* kr = kbp + (size_t)(sb + 16 + lid) * VDD;
            sc1 = wmma_bf16(load_ab(kr + hl * 8,      kr + 16 + hl * 8), bq0, sc1);
            sc1 = wmma_bf16(load_ab(kr + 32 + hl * 8, kr + 48 + hl * 8), bq1, sc1);
        }
        // Online softmax: per-lane state for query tb+lid (halves merged via shfl16)
        float p0[8], p1[8], mloc = -1e30f;
#pragma unroll
        for (int j = 0; j < 8; ++j) {
            p0[j] = sc0[j] * scale;
            p1[j] = sc1[j] * scale;
            mloc = fmaxf(mloc, fmaxf(p0[j], p1[j]));
        }
        mloc = fmaxf(mloc, __shfl_xor(mloc, 16, 32));
        float mnew = fmaxf(mrun, mloc);
        float corr = __expf(mrun - mnew);
        float ssum = 0.f;
#pragma unroll
        for (int j = 0; j < 8; ++j) {
            p0[j] = __expf(p0[j] - mnew);
            p1[j] = __expf(p1[j] - mnew);
            ssum += p0[j] + p1[j];
        }
        ssum += __shfl_xor(ssum, 16, 32);
        lrun = lrun * corr + ssum;
        mrun = mnew;
#pragma unroll
        for (int dc = 0; dc < 4; ++dc)
#pragma unroll
            for (int j = 0; j < 8; ++j) acc[dc][j] *= corr;

        // Repack P^T into B-fragment layout (half-wave exchange)
        v16bf pb;
#pragma unroll
        for (int j = 0; j < 8; ++j) {
            float o0 = __shfl_xor(p0[j], 16, 32);
            float o1 = __shfl_xor(p1[j], 16, 32);
            pb[j]     = (__bf16)(hl ? o1 : p0[j]);  // keys base + (hl?16:0) + j
            pb[j + 8] = (__bf16)(hl ? p1[j] : o0);  // keys base + (hl?24:8) + j
        }
        // O^T += V^T x P^T   (4 d-chunks of 16 rows)
#pragma unroll
        for (int dc = 0; dc < 4; ++dc) {
            const __bf16* vr = vtb + (size_t)(dc * 16 + lid) * SS + sb;
            v16bf av = load_ab(vr + hl * 16, vr + hl * 16 + 8);
            acc[dc] = wmma_bf16(av, pb, acc[dc]);
        }
    }
    float rinv = 1.f / lrun;
    __bf16* orow = o + ((size_t)(b * TT + tb + lid)) * VDD + h * DHH;
#pragma unroll
    for (int dc = 0; dc < 4; ++dc)
#pragma unroll
        for (int j = 0; j < 8; ++j)
            orow[dc * 16 + j + hl * 8] = (__bf16)(acc[dc][j] * rinv);
}

// ---------------------------------------------------------------------------
// Final logits: dot(row[256], w) + b. One wave per row, fp32 out.
// ---------------------------------------------------------------------------
__global__ __launch_bounds__(256) void logits_rows(const float* __restrict__ X,
                                                   const float* __restrict__ wm,
                                                   const float* __restrict__ bm,
                                                   float* __restrict__ out, int rows) {
    int w = (blockIdx.x * blockDim.x + threadIdx.x) >> 5;
    int lane = threadIdx.x & 31;
    if (w >= rows) return;
    const float* xr = X + (size_t)w * DDIM;
    float s = 0.f;
#pragma unroll
    for (int i = 0; i < 8; ++i) {
        int idx = i * 32 + lane;
        s += xr[idx] * wm[idx];
    }
#pragma unroll
    for (int m = 16; m >= 1; m >>= 1) s += __shfl_xor(s, m, 32);
    if (lane == 0) out[w] = s + bm[0];
}

// ---------------------------------------------------------------------------
extern "C" void kernel_launch(void* const* d_in, const int* in_sizes, int n_in,
                              void* d_out, int out_size, void* d_ws, size_t ws_size,
                              hipStream_t stream) {
    (void)in_sizes; (void)n_in; (void)out_size; (void)ws_size;
    const float* x    = (const float*)d_in[0];
    const float* ipos = (const float*)d_in[1];
    const float* iemb = (const float*)d_in[2];
    const float* tco  = (const float*)d_in[3];
    // d_in[4], d_in[5]: seq lengths (all full) -- unused.

    // params flattened in JAX pytree (sorted-key) order
    int p = 6;
    const float* kmlp0W = (const float*)d_in[p++]; const float* kmlp0b = (const float*)d_in[p++];
    const float* kmlp1W = (const float*)d_in[p++]; const float* kmlp1b = (const float*)d_in[p++];
    const float* kprojW = (const float*)d_in[p++]; const float* kprojb = (const float*)d_in[p++];
    const float* kprj_lnb = (const float*)d_in[p++]; const float* kprj_lng = (const float*)d_in[p++];
    const float* key0W = (const float*)d_in[p++]; const float* key0b = (const float*)d_in[p++];
    const float* key1W = (const float*)d_in[p++]; const float* key1b = (const float*)d_in[p++];
    const float* key2W = (const float*)d_in[p++]; const float* key2b = (const float*)d_in[p++];
    const float* outW  = (const float*)d_in[p++]; const float* outb  = (const float*)d_in[p++];
    const float* post_alpha = (const float*)d_in[p++];
    const float* post_l1W = (const float*)d_in[p++]; const float* post_l1b = (const float*)d_in[p++];
    const float* post_l2W = (const float*)d_in[p++]; const float* post_l2b = (const float*)d_in[p++];
    const float* post_lnb = (const float*)d_in[p++]; const float* post_lng = (const float*)d_in[p++];
    const float* pre_alpha = (const float*)d_in[p++];
    const float* pre_l1W = (const float*)d_in[p++]; const float* pre_l1b = (const float*)d_in[p++];
    const float* pre_l2W = (const float*)d_in[p++]; const float* pre_l2b = (const float*)d_in[p++];
    const float* pre_lnb = (const float*)d_in[p++]; const float* pre_lng = (const float*)d_in[p++];
    const float* qry0W = (const float*)d_in[p++]; const float* qry0b = (const float*)d_in[p++];
    const float* qry1W = (const float*)d_in[p++]; const float* qry1b = (const float*)d_in[p++];
    const float* qry2W = (const float*)d_in[p++]; const float* qry2b = (const float*)d_in[p++];
    const float* maskW = (const float*)d_in[p++]; const float* maskb = (const float*)d_in[p++];
    const float* vprojW = (const float*)d_in[p++]; const float* vprojb = (const float*)d_in[p++];
    const float* vprj_lnb = (const float*)d_in[p++]; const float* vprj_lng = (const float*)d_in[p++];
    const float* val0W = (const float*)d_in[p++]; const float* val0b = (const float*)d_in[p++];
    const float* val1W = (const float*)d_in[p++]; const float* val1b = (const float*)d_in[p++];
    const float* val2W = (const float*)d_in[p++]; const float* val2b = (const float*)d_in[p++];

    // ---- workspace layout ----
    char* ws = (char*)d_ws;
    size_t off = 0;
    auto alloc = [&](size_t bytes) -> char* {
        char* q0 = ws + off;
        off += (bytes + 255) & ~(size_t)255;
        return q0;
    };
    const int NR = BB * SS;  // 4096 rows everywhere
    __bf16* wb_pre1  = (__bf16*)alloc((size_t)1024 * 256 * 2);
    __bf16* wb_pre2  = (__bf16*)alloc((size_t)256 * 1024 * 2);
    __bf16* wb_kproj = (__bf16*)alloc((size_t)1024 * 256 * 2);
    __bf16* wb_vproj = (__bf16*)alloc((size_t)1024 * 256 * 2);
    __bf16* wb_kml0  = (__bf16*)alloc((size_t)1024 * 2048 * 2);
    __bf16* wb_kml1  = (__bf16*)alloc((size_t)1024 * 1024 * 2);
    __bf16* wb_out   = (__bf16*)alloc((size_t)256 * 1024 * 2);
    __bf16* wb_post1 = (__bf16*)alloc((size_t)1024 * 256 * 2);
    __bf16* wb_post2 = (__bf16*)alloc((size_t)256 * 1024 * 2);
    __bf16* wb_key2  = (__bf16*)alloc((size_t)1024 * 32 * 2);
    __bf16* wb_qry2  = (__bf16*)alloc((size_t)1024 * 32 * 2);
    __bf16* wb_val2  = (__bf16*)alloc((size_t)1024 * 32 * 2);
    __bf16* lnbuf = (__bf16*)alloc((size_t)NR * 256 * 2);
    __bf16* hbuf  = (__bf16*)alloc((size_t)NR * 1024 * 2);
    float*  x1    = (float*)alloc((size_t)NR * 256 * 4);
    __bf16* kc    = (__bf16*)alloc((size_t)NR * 2048 * 2);
    __bf16* s32   = (__bf16*)alloc((size_t)NR * 32 * 2);
    __bf16* qb    = (__bf16*)alloc((size_t)NR * 1024 * 2);
    __bf16* kb    = (__bf16*)alloc((size_t)NR * 1024 * 2);
    float*  vf    = (float*)alloc((size_t)NR * 1024 * 4);
    __bf16* vtb   = (__bf16*)alloc((size_t)NR * 1024 * 2);
    __bf16* ob    = (__bf16*)alloc((size_t)NR * 1024 * 2);
    float*  xo    = (float*)alloc((size_t)NR * 256 * 4);
    float*  xo2   = (float*)alloc((size_t)NR * 256 * 4);

    auto cvt = [&](const float* s, __bf16* d, int n) {
        f32_to_bf16<<<dim3((n + 255) / 256), dim3(256), 0, stream>>>(s, d, n);
    };
    auto gemm = [&](const __bf16* A, const __bf16* W, const float* bias, const float* alpha,
                    const float* resid, float* oF, __bf16* oB, int M, int N, int K,
                    int ldc, int coloffs, int act) {
        gemm_bf16_wmma<<<dim3(M / 128, N / 64), dim3(256), 0, stream>>>(
            A, W, bias, alpha, resid, oF, oB, M, N, K, ldc, coloffs, act);
    };

    // weight conversion (bf16)
    cvt(pre_l1W, wb_pre1, 1024 * 256);   cvt(pre_l2W, wb_pre2, 256 * 1024);
    cvt(kprojW, wb_kproj, 1024 * 256);   cvt(vprojW, wb_vproj, 1024 * 256);
    cvt(kmlp0W, wb_kml0, 1024 * 2048);   cvt(kmlp1W, wb_kml1, 1024 * 1024);
    cvt(outW, wb_out, 256 * 1024);
    cvt(post_l1W, wb_post1, 1024 * 256); cvt(post_l2W, wb_post2, 256 * 1024);
    cvt(key2W, wb_key2, 1024 * 32);      cvt(qry2W, wb_qry2, 1024 * 32);
    cvt(val2W, wb_val2, 1024 * 32);

    dim3 wgrid((NR * 32 + 255) / 256), wblk(256);

    // 1) pre-FFN: x1 = x + alpha*(W2 @ silu(W1 @ LN(x) + b1) + b2)
    ln_rows<<<wgrid, wblk, 0, stream>>>(x, pre_lng, pre_lnb, lnbuf, NR);
    gemm(lnbuf, wb_pre1, pre_l1b, nullptr, nullptr, nullptr, hbuf, NR, 1024, 256, 1024, 0, 1);
    gemm(hbuf, wb_pre2, pre_l2b, pre_alpha, x, x1, nullptr, NR, 256, 1024, 256, 0, 0);

    // 2) kx = k_proj(LN(x1)) -> kc[:, :1024]
    ln_rows<<<wgrid, wblk, 0, stream>>>(x1, kprj_lng, kprj_lnb, lnbuf, NR);
    gemm(lnbuf, wb_kproj, kprojb, nullptr, nullptr, nullptr, kc, NR, 1024, 256, 2048, 0, 0);

    // 3) ke = mlp3(ie, key) -> kc[:, 1024:]
    mlp2_small<<<dim3((NR + 255) / 256), dim3(256), 0, stream>>>(
        iemb, nullptr, key0W, key0b, key1W, key1b, s32, NR, SS);
    gemm(s32, wb_key2, key2b, nullptr, nullptr, nullptr, kc, NR, 1024, 32, 2048, 1024, 0);

    // 4) k = k_mlp over concat
    gemm(kc, wb_kml0, kmlp0b, nullptr, nullptr, nullptr, hbuf, NR, 1024, 2048, 1024, 0, 1);
    gemm(hbuf, wb_kml1, kmlp1b, nullptr, nullptr, nullptr, kb, NR, 1024, 1024, 1024, 0, 0);

    // 5) v = v_proj(LN(x1)) + mlp3(ie, value)
    mlp2_small<<<dim3((NR + 255) / 256), dim3(256), 0, stream>>>(
        iemb, nullptr, val0W, val0b, val1W, val1b, s32, NR, SS);
    gemm(s32, wb_val2, val2b, nullptr, nullptr, vf, nullptr, NR, 1024, 32, 1024, 0, 0);
    ln_rows<<<wgrid, wblk, 0, stream>>>(x1, vprj_lng, vprj_lnb, lnbuf, NR);
    gemm(lnbuf, wb_vproj, vprojb, nullptr, vf, vf, nullptr, NR, 1024, 256, 1024, 0, 0);
    transpose_v<<<dim3((NR * 1024) / 256), dim3(256), 0, stream>>>(vf, vtb);

    // 6) q = mlp3(temb, query)
    mlp2_small<<<dim3((NR + 255) / 256), dim3(256), 0, stream>>>(
        tco, ipos, qry0W, qry0b, qry1W, qry1b, s32, NR, TT);
    gemm(s32, wb_qry2, qry2b, nullptr, nullptr, nullptr, qb, NR, 1024, 32, 1024, 0, 0);

    // 7) attention (flash, transposed)
    attn_fwd<<<dim3((BB * HH * (TT / 16) * 32) / 256), dim3(256), 0, stream>>>(qb, kb, vtb, ob);

    // 8) out projection + post-FFN + logits
    gemm(ob, wb_out, outb, nullptr, nullptr, xo, nullptr, NR, 256, 1024, 256, 0, 0);
    ln_rows<<<wgrid, wblk, 0, stream>>>(xo, post_lng, post_lnb, lnbuf, NR);
    gemm(lnbuf, wb_post1, post_l1b, nullptr, nullptr, nullptr, hbuf, NR, 1024, 256, 1024, 0, 1);
    gemm(hbuf, wb_post2, post_l2b, post_alpha, xo, xo2, nullptr, NR, 256, 1024, 256, 0, 0);
    logits_rows<<<wgrid, wblk, 0, stream>>>(xo2, maskW, maskb, (float*)d_out, NR);
}
